// WaveletDualStreamVimUNet_40939628266034
// MI455X (gfx1250) — matrix operand order repaired and checked by reading, not compiled, not run on hardware
//
#include <hip/hip_runtime.h>

// ---------------- problem constants ----------------
#define BATCH   2
#define SEQN    4096          // H*W
#define DMODEL  192
#define DINNER  384           // DI
#define TWO_DI  768
#define DSTATE  16
#define DTRANK  12
#define DBCW    44            // DTRANK + 2*DSTATE
#define DCONV   4
#define MROWS   (BATCH*SEQN)  // 8192
#define EPSV    1e-5f

typedef __attribute__((ext_vector_type(16))) __bf16          v16bf;
typedef __attribute__((ext_vector_type(8)))  float           v8f;
typedef __attribute__((ext_vector_type(8)))  unsigned short  v8u;
typedef __attribute__((ext_vector_type(16))) unsigned short  v16u;

__device__ __forceinline__ unsigned short f2bf(float f) {
  unsigned int u = __float_as_uint(f);
  u += 0x7FFFu + ((u >> 16) & 1u);            // round-to-nearest-even
  return (unsigned short)(u >> 16);
}

// ---- WMMA GEMM: C[M,Nc] = A[M,K] * W[Nc,K]^T (+bias) (+=) ----------------
// A, W stored bf16 row-major (as u16). One wave computes a 16x64 output
// strip: one A fragment per K-step feeds 4 WMMAs (A reuse x4).
__global__ void gemm_bf16_wmma(const unsigned short* __restrict__ A,
                               const unsigned short* __restrict__ W,
                               float* __restrict__ C,
                               const float* __restrict__ bias,
                               int M, int K, int Nc, int accumulate)
{
  int wid  = blockIdx.x * (blockDim.x >> 5) + (threadIdx.x >> 5);
  int lane = threadIdx.x & 31;
  int nGroups = Nc >> 6;              // 64-column groups
  int mt = wid / nGroups;
  int ng = wid - mt * nGroups;
  if (mt * 16 >= M) return;
  int half = lane >> 4;               // 0: lanes 0-15, 1: lanes 16-31
  int l16  = lane & 15;

  const unsigned short* Arow = A + (size_t)(mt * 16 + l16) * K;
  const unsigned short* Wr0  = W + (size_t)(ng * 64 + l16) * K;
  const unsigned short* Wr1  = Wr0 + (size_t)16 * K;
  const unsigned short* Wr2  = Wr0 + (size_t)32 * K;
  const unsigned short* Wr3  = Wr0 + (size_t)48 * K;

  v8f acc0 = {0.f,0.f,0.f,0.f,0.f,0.f,0.f,0.f};
  v8f acc1 = acc0, acc2 = acc0, acc3 = acc0;

  for (int kb = 0; kb < K; kb += 32) {
    // A frag: element e -> K=kb+16*half+e
    v8u alo = *(const v8u*)(Arow + kb + 16 * half);
    v8u ahi = *(const v8u*)(Arow + kb + 16 * half + 8);
    v16u au = __builtin_shufflevector(alo, ahi,
              0,1,2,3,4,5,6,7,8,9,10,11,12,13,14,15);
    union { v16u u; v16bf b; } ua, b0, b1, b2, b3;
    ua.u = au;
    // B frag: lanes 0-15 hold K=kb..kb+15, lanes 16-31 hold K=kb+16..kb+31
    b0.u = *(const v16u*)(Wr0 + kb + 16 * half);
    b1.u = *(const v16u*)(Wr1 + kb + 16 * half);
    b2.u = *(const v16u*)(Wr2 + kb + 16 * half);
    b3.u = *(const v16u*)(Wr3 + kb + 16 * half);
    acc0 = __builtin_amdgcn_wmma_f32_16x16x32_bf16(false, ua.b, false, b0.b,
                                                   (short)0, acc0, false, false);
    acc1 = __builtin_amdgcn_wmma_f32_16x16x32_bf16(false, ua.b, false, b1.b,
                                                   (short)0, acc1, false, false);
    acc2 = __builtin_amdgcn_wmma_f32_16x16x32_bf16(false, ua.b, false, b2.b,
                                                   (short)0, acc2, false, false);
    acc3 = __builtin_amdgcn_wmma_f32_16x16x32_bf16(false, ua.b, false, b3.b,
                                                   (short)0, acc3, false, false);
  }

  v8f accs[4] = {acc0, acc1, acc2, acc3};
#pragma unroll
  for (int j = 0; j < 4; ++j) {
    int ncol = ng * 64 + j * 16 + l16;
    float bv = bias ? bias[ncol] : 0.f;
#pragma unroll
    for (int r = 0; r < 8; ++r) {
      int mrow = mt * 16 + r + 8 * half;  // C layout: VGPR r -> M = r + 8*half
      float v = accs[j][r] + bv;
      float* dst = C + (size_t)mrow * Nc + ncol;
      if (accumulate) *dst += v; else *dst = v;
    }
  }
}

// ---------------- elementwise / prep kernels ----------------
__global__ void input_proj_k(const float* __restrict__ x, const float* __restrict__ pw,
                             const float* __restrict__ pb, float* __restrict__ seq)
{
  int t = blockIdx.x * blockDim.x + threadIdx.x;   // MROWS*DMODEL
  int d = t % DMODEL;
  int m = t / DMODEL;
  int b = m >> 12;
  int n = m & (SEQN - 1);
  float acc = pb[d];
#pragma unroll
  for (int c = 0; c < 3; ++c)
    acc += x[((b * 3 + c) << 12) + n] * pw[d * 3 + c];
  seq[t] = acc;
}

__global__ void prep_comb_k(const float* __restrict__ seq, unsigned short* __restrict__ abf)
{
  int t = blockIdx.x * blockDim.x + threadIdx.x;   // MROWS*DINNER
  int c = t % DINNER;
  int m = t / DINNER;
  int n = m & (SEQN - 1);
  float v;
  if (c < DMODEL) {
    v = seq[(size_t)m * DMODEL + c];
  } else {
    int cc = c - DMODEL;
    float cur  = seq[(size_t)m * DMODEL + cc];
    float prev = (n > 0) ? seq[(size_t)(m - 1) * DMODEL + cc] : 0.f;
    v = cur - prev;
  }
  abf[t] = f2bf(v);
}

__global__ void conv_f32_bf16_k(const float* __restrict__ src,
                                unsigned short* __restrict__ dst, int n)
{
  int t = blockIdx.x * blockDim.x + threadIdx.x;
  if (t < n) dst[t] = f2bf(src[t]);
}

__global__ void rmsnorm_bf_k(const float* __restrict__ xin, const float* __restrict__ w,
                             unsigned short* __restrict__ outbf)
{
  int lane = threadIdx.x & 31;
  int row  = blockIdx.x * 8 + (threadIdx.x >> 5);
  const float* xr = xin + (size_t)row * DMODEL;
  float ss = 0.f;
  for (int c = lane; c < DMODEL; c += 32) { float v = xr[c]; ss += v * v; }
#pragma unroll
  for (int o = 16; o > 0; o >>= 1) ss += __shfl_xor(ss, o, 32);
  float inv = rsqrtf(ss * (1.f / DMODEL) + EPSV);
  for (int c = lane; c < DMODEL; c += 32)
    outbf[(size_t)row * DMODEL + c] = f2bf(xr[c] * inv * w[c]);
}

__global__ void conv_silu_k(const float* __restrict__ xz, const float* __restrict__ cw,
                            const float* __restrict__ cb, float* __restrict__ xi)
{
  int t = blockIdx.x * blockDim.x + threadIdx.x;   // MROWS*DINNER
  int i = t % DINNER;
  int m = t / DINNER;
  int n = m & (SEQN - 1);
  float acc = cb[i];
#pragma unroll
  for (int j = 0; j < DCONV; ++j) {
    int nn = n - (DCONV - 1) + j;
    if (nn >= 0)
      acc += cw[i * DCONV + j] * xz[(size_t)(m - (DCONV - 1) + j) * TWO_DI + i];
  }
  xi[t] = acc / (1.f + __expf(-acc));              // silu
}

__global__ void xproj_k(const float* __restrict__ xi, const float* __restrict__ xw,
                        float* __restrict__ dbc)
{
  int t = blockIdx.x * blockDim.x + threadIdx.x;
  if (t >= MROWS * DBCW) return;
  int e = t % DBCW;
  int m = t / DBCW;
  const float* xr = xi + (size_t)m * DINNER;
  const float* wr = xw + (size_t)e * DINNER;
  float acc = 0.f;
  for (int i = 0; i < DINNER; ++i) acc += xr[i] * wr[i];
  dbc[t] = acc;
}

__global__ void dtproj_k(const float* __restrict__ dbc, const float* __restrict__ dw,
                         const float* __restrict__ db, float* __restrict__ dt)
{
  int t = blockIdx.x * blockDim.x + threadIdx.x;   // MROWS*DINNER
  int i = t % DINNER;
  int m = t / DINNER;
  float acc = db[i];
#pragma unroll
  for (int r = 0; r < DTRANK; ++r)
    acc += dbc[(size_t)m * DBCW + r] * dw[i * DTRANK + r];
  dt[t] = (acc > 20.f) ? acc : log1pf(__expf(acc)); // softplus
}

// ---- selective scan: lane = (b,i,s); reduce DS over 16 lanes; fused gate ---
// Lane s==0 of each 16-lane group applies the silu(z) gate and writes the
// bf16 GEMM operand directly (no f32 y round-trip).
__global__ void scan_k(const float* __restrict__ dt, const float* __restrict__ dbc,
                       const float* __restrict__ xi, const float* __restrict__ xz,
                       const float* __restrict__ Alog, const float* __restrict__ Dp,
                       unsigned short* __restrict__ ybf)
{
  int t  = blockIdx.x * blockDim.x + threadIdx.x;  // BATCH*DINNER*DSTATE = 12288
  int s  = t & (DSTATE - 1);
  int ch = t >> 4;
  int b  = ch / DINNER;
  int i  = ch - b * DINNER;
  float Ai  = -__expf(Alog[i * DSTATE + s]);
  float dpv = Dp[i];
  float h = 0.f;
  for (int n = 0; n < SEQN; ++n) {
    int m = (b << 12) + n;
    float dtv = dt[(size_t)m * DINNER + i];
    float xiv = xi[(size_t)m * DINNER + i];
    float bmv = dbc[(size_t)m * DBCW + DTRANK + s];
    float cmv = dbc[(size_t)m * DBCW + DTRANK + DSTATE + s];
    h = __expf(dtv * Ai) * h + (dtv * bmv) * xiv;
    float part = h * cmv;
    part += __shfl_xor(part, 1, 32);
    part += __shfl_xor(part, 2, 32);
    part += __shfl_xor(part, 4, 32);
    part += __shfl_xor(part, 8, 32);
    if (s == 0) {
      float z = xz[(size_t)m * TWO_DI + DINNER + i];
      float v = (part + dpv * xiv) * (z / (1.f + __expf(-z)));
      ybf[(size_t)m * DINNER + i] = f2bf(v);
    }
  }
}

__global__ void final_norm_k(const float* __restrict__ seq, const float* __restrict__ w,
                             float* __restrict__ out)
{
  int lane = threadIdx.x & 31;
  int row  = blockIdx.x * 8 + (threadIdx.x >> 5);
  int b = row >> 12;
  int n = row & (SEQN - 1);
  const float* xr = seq + (size_t)row * DMODEL;
  float ss = 0.f;
  for (int c = lane; c < DMODEL; c += 32) { float v = xr[c]; ss += v * v; }
#pragma unroll
  for (int o = 16; o > 0; o >>= 1) ss += __shfl_xor(ss, o, 32);
  float inv = rsqrtf(ss * (1.f / DMODEL) + EPSV);
  for (int d = lane; d < DMODEL; d += 32)
    out[((size_t)(b * DMODEL + d) << 12) + n] = xr[d] * inv * w[d];
}

// ---------------- host launch ----------------
extern "C" void kernel_launch(void* const* d_in, const int* in_sizes, int n_in,
                              void* d_out, int out_size, void* d_ws, size_t ws_size,
                              hipStream_t stream)
{
  const float* x       = (const float*)d_in[0];
  const float* proj_w  = (const float*)d_in[1];
  const float* proj_b  = (const float*)d_in[2];
  const float* lp_w    = (const float*)d_in[3];
  const float* lp_b    = (const float*)d_in[4];
  const float* norm_w  = (const float*)d_in[5];
  const float* in_w    = (const float*)d_in[6];
  const float* conv_w  = (const float*)d_in[7];
  const float* conv_b  = (const float*)d_in[8];
  const float* xproj_w = (const float*)d_in[9];
  const float* dt_w    = (const float*)d_in[10];
  const float* dt_b    = (const float*)d_in[11];
  const float* A_log   = (const float*)d_in[12];
  const float* Dp      = (const float*)d_in[13];
  const float* out_w   = (const float*)d_in[14];
  const float* normf_w = (const float*)d_in[15];
  float* out = (float*)d_out;

  // workspace layout (all offsets multiples of 256 bytes)
  char* p = (char*)d_ws;
  float* seq   = (float*)p;          p += (size_t)MROWS * DMODEL * 4;
  float* xz    = (float*)p;          p += (size_t)MROWS * TWO_DI * 4;
  float* xibuf = (float*)p;          p += (size_t)MROWS * DINNER * 4;
  float* dtbuf = (float*)p;          p += (size_t)MROWS * DINNER * 4;
  float* dbc   = (float*)p;          p += (size_t)MROWS * DBCW   * 4;
  float* projb = (float*)p;          p += (size_t)MROWS * DMODEL * 4;
  unsigned short* abf  = (unsigned short*)p; p += (size_t)MROWS * DINNER * 2; // comb / gated-y bf16
  unsigned short* xnbf = (unsigned short*)p; p += (size_t)MROWS * DMODEL * 2;
  unsigned short* wbf  = (unsigned short*)p; p += (size_t)TWO_DI * DMODEL * 2;

  input_proj_k<<<MROWS * DMODEL / 256, 256, 0, stream>>>(x, proj_w, proj_b, seq);

  for (int l = 0; l < 4; ++l) {
    // lp projection: comb(bf16) @ lp_w^T + lp_b
    prep_comb_k<<<MROWS * DINNER / 256, 256, 0, stream>>>(seq, abf);
    conv_f32_bf16_k<<<(DMODEL * DINNER + 255) / 256, 256, 0, stream>>>(
        lp_w + (size_t)l * DMODEL * DINNER, wbf, DMODEL * DINNER);
    gemm_bf16_wmma<<<(MROWS / 16) * (DMODEL / 64) / 4, 128, 0, stream>>>(
        abf, wbf, projb, lp_b + l * DMODEL, MROWS, DINNER, DMODEL, 0);

    // rmsnorm -> bf16 activation for in-proj
    rmsnorm_bf_k<<<MROWS / 8, 256, 0, stream>>>(projb, norm_w + l * DMODEL, xnbf);

    // in-proj: xn @ in_w^T -> xz (768)
    conv_f32_bf16_k<<<(TWO_DI * DMODEL + 255) / 256, 256, 0, stream>>>(
        in_w + (size_t)l * TWO_DI * DMODEL, wbf, TWO_DI * DMODEL);
    gemm_bf16_wmma<<<(MROWS / 16) * (TWO_DI / 64) / 4, 128, 0, stream>>>(
        xnbf, wbf, xz, nullptr, MROWS, DMODEL, TWO_DI, 0);

    // causal depthwise conv + silu
    conv_silu_k<<<MROWS * DINNER / 256, 256, 0, stream>>>(
        xz, conv_w + l * DINNER * DCONV, conv_b + l * DINNER, xibuf);

    // dbc = xi @ xproj_w^T ; dt = softplus(dbc[:,:12] @ dt_w^T + dt_b)
    xproj_k<<<(MROWS * DBCW + 255) / 256, 256, 0, stream>>>(
        xibuf, xproj_w + (size_t)l * DBCW * DINNER, dbc);
    dtproj_k<<<MROWS * DINNER / 256, 256, 0, stream>>>(
        dbc, dt_w + (size_t)l * DINNER * DTRANK, dt_b + l * DINNER, dtbuf);

    // selective scan with fused +Dp*xi, silu(z) gate, bf16 store
    scan_k<<<(BATCH * DINNER * DSTATE) / 256, 256, 0, stream>>>(
        dtbuf, dbc, xibuf, xz, A_log + (size_t)l * DINNER * DSTATE,
        Dp + l * DINNER, abf);

    // out-proj with fused residual into seq
    conv_f32_bf16_k<<<(DMODEL * DINNER + 255) / 256, 256, 0, stream>>>(
        out_w + (size_t)l * DMODEL * DINNER, wbf, DMODEL * DINNER);
    gemm_bf16_wmma<<<(MROWS / 16) * (DMODEL / 64) / 4, 128, 0, stream>>>(
        abf, wbf, seq, nullptr, MROWS, DINNER, DMODEL, 1);
  }

  final_norm_k<<<MROWS / 8, 256, 0, stream>>>(seq, normf_w, out);
}
